// Model_HyperCluster_61838939128326
// MI455X (gfx1250) — matrix operand order repaired and checked by reading, not compiled, not run on hardware
//
#include <hip/hip_runtime.h>
#include <hip/hip_bf16.h>
#include <math.h>

// ---- static model sizes ----
#define NE    131072        // NUM_EDGES (dual nodes)
#define NH    163840        // NUM_HEDGES
#define HID   256
#define FE    128
#define BG    64
#define SEEDS 512
#define EPG   2048

typedef __attribute__((ext_vector_type(16))) __bf16 v16bf;
typedef __attribute__((ext_vector_type(8)))  float  v8f;

union Frag { uint4 u[2]; v16bf v; };
union Pack { __bf16 h[16]; uint4 u[2]; };

#define FLAG_TA   1
#define FLAG_TB   2
#define FLAG_RELU 4

#define BM 128
#define BN 128
#define BK 32
#define LDT 40   // bf16 elements per LDS row (80B, 16B-aligned rows)

// ---------------------------------------------------------------------------
// Generic WMMA bf16 GEMM: C = act(alpha * op(A) @ op(B) + bias + addsrc)
// grid = (N/128, M/128, batch). A,B fp32 in memory, converted to bf16 in LDS.
// ---------------------------------------------------------------------------
__global__ __launch_bounds__(256) void wmma_gemm_k(
    const float* __restrict__ A, const float* __restrict__ B, float* __restrict__ C,
    const float* __restrict__ bias, const float* __restrict__ addsrc,
    int K, int lda, int ldb, int ldc,
    long long sA, long long sB, long long sC, long long sAdd,
    float alpha, int flags)
{
    __shared__ __align__(16) __bf16 As[BM][LDT];
    __shared__ __align__(16) __bf16 Bs[BN][LDT];

    A += (long long)blockIdx.z * sA;
    B += (long long)blockIdx.z * sB;
    C += (long long)blockIdx.z * sC;
    if (addsrc) addsrc += (long long)blockIdx.z * sAdd;

    const int tid  = threadIdx.x;
    const int lane = tid & 31;
    const int wave = tid >> 5;
    const int wm   = (wave & 3) * 32;   // wave M offset within tile
    const int wn   = (wave >> 2) * 64;  // wave N offset within tile
    const int m0   = blockIdx.y * BM;
    const int n0   = blockIdx.x * BN;
    const int hi   = lane >> 4;
    const int lr   = lane & 15;

    v8f acc[2][4];
    const v8f vzero = {0.f,0.f,0.f,0.f,0.f,0.f,0.f,0.f};
#pragma unroll
    for (int mi = 0; mi < 2; ++mi)
#pragma unroll
        for (int ni = 0; ni < 4; ++ni) acc[mi][ni] = vzero;

    for (int k0 = 0; k0 < K; k0 += BK) {
        __syncthreads();
        // ---- stage A tile -> As[m][k] (bf16) ----
        if (!(flags & FLAG_TA)) {
            int m  = tid >> 1;
            int cc = (tid & 1) * 16;
            const float4* g = (const float4*)(A + (long long)(m0 + m) * lda + k0 + cc);
            Pack pk;
#pragma unroll
            for (int j = 0; j < 4; ++j) {
                float4 f = g[j];
                pk.h[4*j + 0] = (__bf16)f.x; pk.h[4*j + 1] = (__bf16)f.y;
                pk.h[4*j + 2] = (__bf16)f.z; pk.h[4*j + 3] = (__bf16)f.w;
            }
            *(uint4*)&As[m][cc]     = pk.u[0];   // packed cvt + ds_store_b128
            *(uint4*)&As[m][cc + 8] = pk.u[1];
        } else { // A supplied K-major: element op(A)[m][k] = A[k][m]
            int k  = tid >> 3;
            int mc = (tid & 7) * 16;
            const float4* g = (const float4*)(A + (long long)(k0 + k) * lda + m0 + mc);
#pragma unroll
            for (int j = 0; j < 4; ++j) {
                float4 f = g[j];
                As[mc + 4*j + 0][k] = (__bf16)f.x;
                As[mc + 4*j + 1][k] = (__bf16)f.y;
                As[mc + 4*j + 2][k] = (__bf16)f.z;
                As[mc + 4*j + 3][k] = (__bf16)f.w;
            }
        }
        // ---- stage B tile -> Bs[n][k] (bf16, i.e. B^T layout) ----
        if (!(flags & FLAG_TB)) { // B is K x N row-major (weights, V)
            int k  = tid >> 3;
            int nc = (tid & 7) * 16;
            const float4* g = (const float4*)(B + (long long)(k0 + k) * ldb + n0 + nc);
#pragma unroll
            for (int j = 0; j < 4; ++j) {
                float4 f = g[j];
                Bs[nc + 4*j + 0][k] = (__bf16)f.x;
                Bs[nc + 4*j + 1][k] = (__bf16)f.y;
                Bs[nc + 4*j + 2][k] = (__bf16)f.z;
                Bs[nc + 4*j + 3][k] = (__bf16)f.w;
            }
        } else { // B supplied as N x K row-major (e.g. K-matrix for Q@K^T)
            int n  = tid >> 1;
            int cc = (tid & 1) * 16;
            const float4* g = (const float4*)(B + (long long)(n0 + n) * ldb + k0 + cc);
            Pack pk;
#pragma unroll
            for (int j = 0; j < 4; ++j) {
                float4 f = g[j];
                pk.h[4*j + 0] = (__bf16)f.x; pk.h[4*j + 1] = (__bf16)f.y;
                pk.h[4*j + 2] = (__bf16)f.z; pk.h[4*j + 3] = (__bf16)f.w;
            }
            *(uint4*)&Bs[n][cc]     = pk.u[0];
            *(uint4*)&Bs[n][cc + 8] = pk.u[1];
        }
        __syncthreads();

        // ---- fragment gather per ISA 7.12.2 (16-bit A 16x32 layout) ----
        Frag a[2], b[4];
#pragma unroll
        for (int mi = 0; mi < 2; ++mi) {
            const __bf16* r = &As[wm + mi*16 + lr][0];
            a[mi].u[0] = *(const uint4*)(r + 8*hi);        // K = 8*hi .. 8*hi+7
            a[mi].u[1] = *(const uint4*)(r + 16 + 8*hi);   // K = 16+8*hi .. +7
        }
#pragma unroll
        for (int ni = 0; ni < 4; ++ni) {
            const __bf16* r = &Bs[wn + ni*16 + lr][0];
            b[ni].u[0] = *(const uint4*)(r + 8*hi);        // K = 8*hi .. 8*hi+7
            b[ni].u[1] = *(const uint4*)(r + 16 + 8*hi);   // K = 16+8*hi .. +7
        }
#pragma unroll
        for (int mi = 0; mi < 2; ++mi)
#pragma unroll
            for (int ni = 0; ni < 4; ++ni)
                acc[mi][ni] = __builtin_amdgcn_wmma_f32_16x16x32_bf16(
                    false, a[mi].v, false, b[ni].v, (short)0, acc[mi][ni], false, false);
    }

    // ---- epilogue: C 16x16 f32 layout: VGPR r, lane-half hi -> row r+8*hi ----
#pragma unroll
    for (int mi = 0; mi < 2; ++mi) {
#pragma unroll
        for (int ni = 0; ni < 4; ++ni) {
            int colg = n0 + wn + ni*16 + lr;
#pragma unroll
            for (int r = 0; r < 8; ++r) {
                int rowg = m0 + wm + mi*16 + hi*8 + r;
                float v = acc[mi][ni][r] * alpha;
                if (bias)   v += bias[colg];
                if (addsrc) v += addsrc[(long long)rowg * ldc + colg];
                if (flags & FLAG_RELU) v = fmaxf(v, 0.f);
                C[(long long)rowg * ldc + colg] = v;
            }
        }
    }
}

// ---------------------------------------------------------------------------
// Scatter / elementwise helpers
// ---------------------------------------------------------------------------
__global__ void fill_zero4_k(float4* __restrict__ p, int n4) {
    int i = blockIdx.x * 256 + threadIdx.x;
    if (i < n4) p[i] = make_float4(0.f, 0.f, 0.f, 0.f);
}

__global__ void count_deg_k(const int* __restrict__ hn, const int* __restrict__ he,
                            float* __restrict__ D, float* __restrict__ Bd, int nnz) {
    int i = blockIdx.x * 256 + threadIdx.x;
    if (i < nnz) {
        atomicAdd(&D[hn[i]], 1.0f);
        atomicAdd(&Bd[he[i]], 1.0f);
    }
}

__global__ void invert_k(float* __restrict__ p, int n) {
    int i = blockIdx.x * 256 + threadIdx.x;
    if (i < n) { float x = p[i]; p[i] = (x > 0.f) ? (1.0f / x) : 0.f; }
}

// m[he[j]] += t[hn[j]]   (channel-parallel float4, f32 atomics -> L2)
__global__ void scat_n2h_k(const float* __restrict__ t, float* __restrict__ m,
                           const int* __restrict__ hn, const int* __restrict__ he,
                           int nnz, int shift) {
    long long gid = (long long)blockIdx.x * 256 + threadIdx.x;
    int entry = (int)(gid >> shift);
    if (entry >= nnz) return;
    int C  = 4 << shift;
    int c4 = (int)(gid & ((1 << shift) - 1)) << 2;
    float4 v = *(const float4*)(t + (long long)hn[entry] * C + c4);
    float* p = m + (long long)he[entry] * C + c4;
    atomicAdd(p + 0, v.x); atomicAdd(p + 1, v.y);
    atomicAdd(p + 2, v.z); atomicAdd(p + 3, v.w);
}

// out[hn[j]] += m[he[j]] * Binv[he[j]]
__global__ void scat_h2n_k(const float* __restrict__ m, float* __restrict__ out,
                           const int* __restrict__ hn, const int* __restrict__ he,
                           const float* __restrict__ Binv, int nnz, int shift) {
    long long gid = (long long)blockIdx.x * 256 + threadIdx.x;
    int entry = (int)(gid >> shift);
    if (entry >= nnz) return;
    int C  = 4 << shift;
    int c4 = (int)(gid & ((1 << shift) - 1)) << 2;
    int h  = he[entry];
    float s = Binv[h];
    float4 v = *(const float4*)(m + (long long)h * C + c4);
    float* p = out + (long long)hn[entry] * C + c4;
    atomicAdd(p + 0, v.x * s); atomicAdd(p + 1, v.y * s);
    atomicAdd(p + 2, v.z * s); atomicAdd(p + 3, v.w * s);
}

// out = out * Dinv[row] + bias[col]  (optional relu)
__global__ void finalize_k(float* __restrict__ out, const float* __restrict__ Dinv,
                           const float* __restrict__ bias, int rows, int shift, int relu) {
    long long gid = (long long)blockIdx.x * 256 + threadIdx.x;
    int row = (int)(gid >> shift);
    if (row >= rows) return;
    int C  = 4 << shift;
    int c4 = (int)(gid & ((1 << shift) - 1)) << 2;
    float d = Dinv[row];
    float* p = out + (long long)row * C + c4;
    float4 v = *(const float4*)p;
    v.x = v.x * d + bias[c4 + 0]; v.y = v.y * d + bias[c4 + 1];
    v.z = v.z * d + bias[c4 + 2]; v.w = v.w * d + bias[c4 + 3];
    if (relu) {
        v.x = fmaxf(v.x, 0.f); v.y = fmaxf(v.y, 0.f);
        v.z = fmaxf(v.z, 0.f); v.w = fmaxf(v.w, 0.f);
    }
    *(float4*)p = v;
}

// softmax over the seed axis: A[b][s][e], reduce over s (stride EPG).
// Online max/sum (one read pass), then one normalize pass (read+write).
__global__ void softmax_seeds_k(float* __restrict__ A) {
    int gid = blockIdx.x * 256 + threadIdx.x;      // BG*EPG threads
    if (gid >= BG * EPG) return;
    int b = gid >> 11, e = gid & (EPG - 1);
    float* base = A + (long long)b * SEEDS * EPG + e;
    float mx = -3.4e38f, sum = 0.f;
    for (int s = 0; s < SEEDS; ++s) {
        float x = base[(long long)s * EPG];
        if (x > mx) { sum = sum * __expf(mx - x) + 1.0f; mx = x; }
        else        { sum += __expf(x - mx); }
    }
    float inv = 1.0f / sum;
    for (int s = 0; s < SEEDS; ++s) {
        float x = base[(long long)s * EPG];
        base[(long long)s * EPG] = __expf(x - mx) * inv;
    }
}

// layernorm over rows of 256: one wave32 per row, shuffle reductions
__global__ void layernorm_k(float* __restrict__ x, const float* __restrict__ g,
                            const float* __restrict__ b, int rows) {
    int row  = (blockIdx.x * 256 + threadIdx.x) >> 5;
    int lane = threadIdx.x & 31;
    if (row >= rows) return;
    float* p = x + (long long)row * HID;
    float v[8]; float s = 0.f;
#pragma unroll
    for (int i = 0; i < 8; ++i) { v[i] = p[lane + 32 * i]; s += v[i]; }
#pragma unroll
    for (int o = 16; o > 0; o >>= 1) s += __shfl_xor(s, o, 32);
    float mu = s * (1.0f / HID);
    float var = 0.f;
#pragma unroll
    for (int i = 0; i < 8; ++i) { float d = v[i] - mu; var += d * d; }
#pragma unroll
    for (int o = 16; o > 0; o >>= 1) var += __shfl_xor(var, o, 32);
    float r = rsqrtf(var * (1.0f / HID) + 1e-5f);
#pragma unroll
    for (int i = 0; i < 8; ++i) {
        int c = lane + 32 * i;
        p[c] = (v[i] - mu) * r * g[c] + b[c];
    }
}

__global__ void ew_add_k(float4* __restrict__ a, const float4* __restrict__ b, int n4) {
    int i = blockIdx.x * 256 + threadIdx.x;
    if (i < n4) {
        float4 x = a[i], y = b[i];
        x.x += y.x; x.y += y.y; x.z += y.z; x.w += y.w;
        a[i] = x;
    }
}

// ---------------------------------------------------------------------------
// Host-side orchestration
// ---------------------------------------------------------------------------
static void launch_gemm(hipStream_t st, const float* A, const float* B, float* C,
                        const float* bias, const float* addsrc,
                        int M, int N, int K, int lda, int ldb, int ldc,
                        long long sA, long long sB, long long sC, long long sAdd,
                        float alpha, int flags, int batch) {
    dim3 grid(N / BN, M / BM, batch), blk(256);
    hipLaunchKernelGGL(wmma_gemm_k, grid, blk, 0, st, A, B, C, bias, addsrc,
                       K, lda, ldb, ldc, sA, sB, sC, sAdd, alpha, flags);
}

static void hconv(hipStream_t st, const float* x, const float* W, const float* bias,
                  int Kdim, int Ndim, float* out, bool relu,
                  float* bufT, float* bufM, const int* hn, const int* he, int nnz,
                  const float* Dinv, const float* Binv) {
    // t = x @ W  (WMMA)
    launch_gemm(st, x, W, bufT, nullptr, nullptr, NE, Ndim, Kdim,
                Kdim, Ndim, Ndim, 0, 0, 0, 0, 1.0f, 0, 1);
    int shift = (Ndim == 256) ? 6 : 5;
    // zero hyperedge accumulator, scatter node->hedge
    int n4m = NH * Ndim / 4;
    hipLaunchKernelGGL(fill_zero4_k, dim3((n4m + 255) / 256), dim3(256), 0, st, (float4*)bufM, n4m);
    long long tot = (long long)nnz << shift;
    int blks = (int)((tot + 255) / 256);
    hipLaunchKernelGGL(scat_n2h_k, dim3(blks), dim3(256), 0, st, bufT, bufM, hn, he, nnz, shift);
    // zero output, scatter hedge->node with Binv scale
    int n4o = NE * Ndim / 4;
    hipLaunchKernelGGL(fill_zero4_k, dim3((n4o + 255) / 256), dim3(256), 0, st, (float4*)out, n4o);
    hipLaunchKernelGGL(scat_h2n_k, dim3(blks), dim3(256), 0, st, bufM, out, hn, he, Binv, nnz, shift);
    // out = out * Dinv + bias (+relu)
    long long fin = (long long)NE << shift;
    hipLaunchKernelGGL(finalize_k, dim3((int)((fin + 255) / 256)), dim3(256), 0, st,
                       out, Dinv, bias, NE, shift, relu ? 1 : 0);
}

extern "C" void kernel_launch(void* const* d_in, const int* in_sizes, int n_in,
                              void* d_out, int out_size, void* d_ws, size_t ws_size,
                              hipStream_t stream) {
    const float* edge_attr = (const float*)d_in[0];
    const float* conv0_W   = (const float*)d_in[1];
    const float* conv0_b   = (const float*)d_in[2];
    const float* convs_W   = (const float*)d_in[3];
    const float* convs_b   = (const float*)d_in[4];
    const float* S         = (const float*)d_in[5];
    const float* q_W = (const float*)d_in[6];  const float* q_b = (const float*)d_in[7];
    const float* k_W = (const float*)d_in[8];  const float* k_b = (const float*)d_in[9];
    const float* v_W = (const float*)d_in[10]; const float* v_b = (const float*)d_in[11];
    const float* o_W = (const float*)d_in[12]; const float* o_b = (const float*)d_in[13];
    const float* ln0_g = (const float*)d_in[14]; const float* ln0_b = (const float*)d_in[15];
    const float* ln1_g = (const float*)d_in[16]; const float* ln1_b = (const float*)d_in[17];
    const float* un_W = (const float*)d_in[18]; const float* un_b = (const float*)d_in[19];
    const float* last_W = (const float*)d_in[20]; const float* last_b = (const float*)d_in[21];
    const int*  hidx = (const int*)d_in[22];
    const int   nnz  = in_sizes[22] / 2;
    const int*  hn   = hidx;
    const int*  he   = hidx + nnz;

    // ---- workspace layout (fp32 elements) ----
    float* w    = (float*)d_ws;
    float* Dinv = w;
    float* Binv = Dinv + NE;
    float* bufT = Binv + NH;
    float* bufM = bufT + (size_t)NE * HID;
    float* bufE = bufM + (size_t)NH * HID;
    float* bufK = bufE + (size_t)NE * HID;
    float* bufV = bufK + (size_t)NE * HID;
    float* bufQ = bufV + (size_t)NE * HID;
    float* bufO = bufQ + (size_t)SEEDS * HID;
    float* bufH = bufO + (size_t)BG * SEEDS * HID;
    float* bufA = bufH + (size_t)BG * SEEDS * HID;
    float* outp = (float*)d_out;
    const float SCALE = 0.0625f; // 1/sqrt(256)

    // ---- degree inverses (shared by every hconv) ----
    {
        int ndeg = NE + NH; // Dinv and Binv contiguous
        hipLaunchKernelGGL(fill_zero4_k, dim3((ndeg / 4 + 255) / 256), dim3(256), 0, stream,
                           (float4*)Dinv, ndeg / 4);
        hipLaunchKernelGGL(count_deg_k, dim3((nnz + 255) / 256), dim3(256), 0, stream,
                           hn, he, Dinv, Binv, nnz);
        hipLaunchKernelGGL(invert_k, dim3((ndeg + 255) / 256), dim3(256), 0, stream, Dinv, ndeg);
    }

    // ---- conv stack: e = relu(hconv(...)) x3 ----
    hconv(stream, edge_attr, conv0_W, conv0_b, FE, HID, bufE, true,
          bufT, bufM, hn, he, nnz, Dinv, Binv);
    hconv(stream, bufE, convs_W + 0 * HID * HID, convs_b + 0 * HID, HID, HID, bufK, true,
          bufT, bufM, hn, he, nnz, Dinv, Binv);
    hconv(stream, bufK, convs_W + 1 * HID * HID, convs_b + 1 * HID, HID, HID, bufE, true,
          bufT, bufM, hn, he, nnz, Dinv, Binv);

    // ---- K, V projections (hconv, no relu) ----
    hconv(stream, bufE, k_W, k_b, HID, HID, bufK, false, bufT, bufM, hn, he, nnz, Dinv, Binv);
    hconv(stream, bufE, v_W, v_b, HID, HID, bufV, false, bufT, bufM, hn, he, nnz, Dinv, Binv);

    // ---- Q = S @ q_W + q_b (batch-invariant: compute once) ----
    launch_gemm(stream, S, q_W, bufQ, q_b, nullptr, SEEDS, HID, HID,
                HID, HID, HID, 0, 0, 0, 0, 1.0f, 0, 1);

    // ---- logits[b] = SCALE * Q @ K[b]^T  (transB: K[b] is N x K row-major) ----
    launch_gemm(stream, bufQ, bufK, bufA, nullptr, nullptr, SEEDS, EPG, HID,
                HID, HID, EPG,
                0, (long long)EPG * HID, (long long)SEEDS * EPG, 0,
                SCALE, FLAG_TB, BG);

    // ---- softmax over seed axis ----
    hipLaunchKernelGGL(softmax_seeds_k, dim3((BG * EPG + 255) / 256), dim3(256), 0, stream, bufA);

    // ---- out[b] = Q + A[b] @ V[b] ----
    launch_gemm(stream, bufA, bufV, bufO, nullptr, bufQ, SEEDS, HID, EPG,
                EPG, HID, HID,
                (long long)SEEDS * EPG, (long long)EPG * HID, (long long)SEEDS * HID, 0,
                1.0f, 0, BG);

    // ---- LN0; out += relu(out @ o_W + o_b); LN1 ----
    int rows = BG * SEEDS;
    hipLaunchKernelGGL(layernorm_k, dim3(rows / 8), dim3(256), 0, stream, bufO, ln0_g, ln0_b, rows);
    launch_gemm(stream, bufO, o_W, bufH, o_b, nullptr, rows, HID, HID,
                HID, HID, HID, 0, 0, 0, 0, 1.0f, FLAG_RELU, 1);
    int n4 = rows * HID / 4;
    hipLaunchKernelGGL(ew_add_k, dim3((n4 + 255) / 256), dim3(256), 0, stream,
                       (float4*)bufO, (const float4*)bufH, n4);
    hipLaunchKernelGGL(layernorm_k, dim3(rows / 8), dim3(256), 0, stream, bufO, ln1_g, ln1_b, rows);

    // ---- e2[b] = A[b]^T @ out[b]  (transA: read A[k][m], lda = EPG) ----
    launch_gemm(stream, bufA, bufO, bufE, nullptr, nullptr, EPG, HID, SEEDS,
                EPG, HID, HID,
                (long long)SEEDS * EPG, (long long)SEEDS * HID, (long long)EPG * HID, 0,
                1.0f, FLAG_TA, BG);

    // ---- un-conv stack: relu(hconv) x3 ----
    hconv(stream, bufE, un_W + 0 * HID * HID, un_b + 0 * HID, HID, HID, bufK, true,
          bufT, bufM, hn, he, nnz, Dinv, Binv);
    hconv(stream, bufK, un_W + 1 * HID * HID, un_b + 1 * HID, HID, HID, bufE, true,
          bufT, bufM, hn, he, nnz, Dinv, Binv);
    hconv(stream, bufE, un_W + 2 * HID * HID, un_b + 2 * HID, HID, HID, bufK, true,
          bufT, bufM, hn, he, nnz, Dinv, Binv);

    // ---- last hconv -> d_out (no relu, N = 128) ----
    hconv(stream, bufK, last_W, last_b, HID, FE, outp, false,
          bufT, bufM, hn, he, nnz, Dinv, Binv);
}
